// DKVMNModel_42348377538635
// MI455X (gfx1250) — matrix operand order repaired and checked by reading, not compiled, not run on hardware
//
#include <hip/hip_runtime.h>
#include <hip/hip_bf16.h>
#include <math.h>

// ---------------------------------------------------------------------------
// DKVMN on MI455X (gfx1250): wave32, WMMA bf16, 320KB LDS-resident value
// memory, async global->LDS double-buffered operand prefetch, TDM bulk copies.
// ---------------------------------------------------------------------------
#define Bsz 2048
#define Ssz 200
#define Kslots 64
#define Vdim 256
#define Ddim 128
#define BPB 4            // batches per scan block (vm: 4*64KB = 256KB LDS)

typedef __attribute__((ext_vector_type(16))) __bf16 v16bf;
typedef __attribute__((ext_vector_type(8)))  float  v8f;
typedef __attribute__((ext_vector_type(4)))  unsigned v4u;
typedef __attribute__((ext_vector_type(8)))  unsigned v8u;

__device__ __forceinline__ v8f wmma_bf16(v16bf a, v16bf b, v8f c) {
  return __builtin_amdgcn_wmma_f32_16x16x32_bf16(false, a, false, b, (short)0, c,
                                                 false, false);
}

// A-operand (16x32 bf16, MxK) from a row-major LDS bf16 buffer.
// lane L<16 -> row L, K = kt*32 + {0..7, 16..23};
// lane L>=16 -> row L-16, K = kt*32 + {8..15, 24..31}.  Two 16B LDS chunks.
__device__ __forceinline__ v16bf lds_a_tile(const __bf16* buf, int rowStride,
                                            int kt, int lane) {
  int row  = lane & 15;
  int koff = (lane & 16) ? 8 : 0;
  const __bf16* p = buf + row * rowStride + kt * 32 + koff;
  union { v16bf v; uint4 q[2]; } u;
  u.q[0] = *(const uint4*)(p);
  u.q[1] = *(const uint4*)(p + 16);
  return u.v;
}

// B-operand (32x16 bf16, KxN) from pre-swizzled global buffer: tile = 512
// contiguous bf16, lane reads its 16 elements contiguously (2x b128).
__device__ __forceinline__ v16bf glb_b_tile(const __bf16* buf, int tileIdx,
                                            int lane) {
  const __bf16* p = buf + ((size_t)tileIdx << 9) + (lane << 4);
  union { v16bf v; uint4 q[2]; } u;
  u.q[0] = *(const uint4*)(p);
  u.q[1] = *(const uint4*)(p + 8);
  return u.v;
}

__device__ __forceinline__ float sigmoidf_(float x) { return 1.f / (1.f + expf(-x)); }

// ---- CDNA5 async global->LDS (ASYNCcnt) ----
__device__ __forceinline__ void async_copy_b128(unsigned lds_off, const void* g) {
  asm volatile("global_load_async_to_lds_b128 %0, %1, off"
               :: "v"(lds_off), "v"((unsigned long long)(uintptr_t)g) : "memory");
}
__device__ __forceinline__ void async_copy_b32(unsigned lds_off, const void* g) {
  asm volatile("global_load_async_to_lds_b32 %0, %1, off"
               :: "v"(lds_off), "v"((unsigned long long)(uintptr_t)g) : "memory");
}
__device__ __forceinline__ void wait_async0() {
  asm volatile("s_wait_asynccnt 0" ::: "memory");
}

// ---- CDNA5 Tensor Data Mover: 1-D f32 tile (n <= 65535 elements) ----
// D# group0: count=1 | lds_addr | global_addr | type=2.  group1: data_size=4B,
// tensor_dim0 = tile_dim0 = n, tensor_dim1 = tile_dim1 = 1, dim0_stride = n.
__device__ __forceinline__ void tdm_1d_f32(unsigned lds_off, const void* gptr,
                                           unsigned n, bool store) {
  unsigned long long ga = (unsigned long long)(uintptr_t)gptr;
  v4u g0 = { 1u, lds_off, (unsigned)ga,
             (unsigned)((ga >> 32) & 0x1FFFFFFu) | (2u << 30) };
  v8u g1 = { 0x20000u,                          // data_size = 2 -> 4 bytes
             (n & 0xFFFFu) << 16,               // tensor_dim0[15:0]  (bits 63:48)
             ((n >> 16) & 0xFFFFu) | (1u << 16),// tensor_dim0[31:16] | tensor_dim1=1
             (n & 0xFFFFu) << 16,               // tile_dim0          (bits 127:112)
             1u,                                // tile_dim1 = 1
             n,                                 // tensor_dim0_stride[31:0]
             0u, 0u };
  if (store)
    asm volatile("tensor_store_from_lds %0, %1" :: "s"(g0), "s"(g1) : "memory");
  else
    asm volatile("tensor_load_to_lds %0, %1"    :: "s"(g0), "s"(g1) : "memory");
}

// ---------------------------------------------------------------------------
// Phase 0: swizzle weights into bf16 B-operand tile layout.
// ---------------------------------------------------------------------------
__global__ void __launch_bounds__(256) dkvmn_prep(
    const float* __restrict__ Wq, const float* __restrict__ key_memory,
    const float* __restrict__ We, const float* __restrict__ Wa,
    const float* __restrict__ Ws,
    __bf16* Wq_sw, __bf16* KT_sw, __bf16* We_sw, __bf16* Wa_sw, __bf16* Ws_sw) {
  int gid = blockIdx.x * 256 + threadIdx.x;
  const float* src; __bf16* dst; int numKt, C, o; bool tr = false;
  if      (gid <  32768) { src = Wq;         dst = Wq_sw; numKt = 4;  C = 256; o = gid; }
  else if (gid <  49152) { src = key_memory; dst = KT_sw; numKt = 8;  C = 64;  o = gid - 32768; tr = true; }
  else if (gid < 114688) { src = We;         dst = We_sw; numKt = 8;  C = 256; o = gid - 49152; }
  else if (gid < 180224) { src = Wa;         dst = Wa_sw; numKt = 8;  C = 256; o = gid - 114688; }
  else if (gid < 229376) { src = Ws;         dst = Ws_sw; numKt = 12; C = 128; o = gid - 180224; }
  else return;
  int tile = o >> 9, within = o & 511, ln = within >> 4, e = within & 15;
  int nt = tile / numKt, kt = tile - nt * numKt;
  int r = kt * 32 + ((e & 8) ? 16 : 0) + ((ln & 16) ? 8 : 0) + (e & 7);   // K index
  int c = nt * 16 + (ln & 15);                  // N index
  float v = tr ? src[c * 256 + r] : src[r * C + c];  // KT: key_memory^T
  dst[o] = (__bf16)v;
}

// ---------------------------------------------------------------------------
// Phase 1: attention weights.  4 waves/block, one 16-row (b,t) tile per wave.
// ---------------------------------------------------------------------------
__global__ void __launch_bounds__(128) dkvmn_attn(
    const int* __restrict__ concept_ids, const float* __restrict__ concept_embed,
    const __bf16* __restrict__ Wq_sw, const __bf16* __restrict__ KT_sw,
    const float* __restrict__ bq, float* __restrict__ w_all) {
  extern __shared__ char smem_raw[];
  const int tid = threadIdx.x, lane = tid & 31, wv = tid >> 5;
  __bf16* embB = (__bf16*)smem_raw;            // [4][16][128] bf16
  __bf16* qB   = embB + 4 * 16 * 128;          // [4][16][256] bf16
  float*  lg   = (float*)(qB + 4 * 16 * 256);  // [4][16][64]  f32
  __bf16* embW = embB + wv * 16 * 128;
  __bf16* qW   = qB + wv * 16 * 256;
  float*  lgW  = lg + wv * 16 * 64;
  const int r0 = (blockIdx.x * 4 + wv) * 16;   // global row = b*S + t

  for (int i = lane; i < 16 * 128; i += 32) {
    int row = i >> 7, d = i & 127;
    int cid = concept_ids[r0 + row];
    embW[i] = (__bf16)concept_embed[(size_t)cid * 128 + d];
  }
  __syncthreads();

  // query (16x256) = emb (16x128) @ Wq (128x256)
  for (int nt = 0; nt < 16; nt++) {
    v8f acc = {};
#pragma unroll
    for (int kt = 0; kt < 4; kt++) {
      v16bf a = lds_a_tile(embW, 128, kt, lane);
      v16bf b = glb_b_tile(Wq_sw, nt * 4 + kt, lane);
      acc = wmma_bf16(a, b, acc);
    }
    int col = nt * 16 + (lane & 15);
    float bias = bq[col];
    int rb = (lane & 16) ? 8 : 0;
#pragma unroll
    for (int vr = 0; vr < 8; vr++)
      qW[(vr + rb) * 256 + col] = (__bf16)(acc[vr] + bias);
  }
  __syncthreads();

  // logits (16x64) = query (16x256) @ key_memory^T (256x64)
  for (int nt = 0; nt < 4; nt++) {
    v8f acc = {};
#pragma unroll
    for (int kt = 0; kt < 8; kt++) {
      v16bf a = lds_a_tile(qW, 256, kt, lane);
      v16bf b = glb_b_tile(KT_sw, nt * 8 + kt, lane);
      acc = wmma_bf16(a, b, acc);
    }
    int col = nt * 16 + (lane & 15);
    int rb = (lane & 16) ? 8 : 0;
#pragma unroll
    for (int vr = 0; vr < 8; vr++) lgW[(vr + rb) * 64 + col] = acc[vr];
  }
  __syncthreads();

  if (lane < 16) {
    float* row = lgW + lane * 64;
    float mx = row[0];
    for (int k = 1; k < 64; k++) mx = fmaxf(mx, row[k]);
    float s = 0.f;
    for (int k = 0; k < 64; k++) { float ev = expf(row[k] - mx); row[k] = ev; s += ev; }
    float inv = 1.f / s;
    size_t base = (size_t)(r0 + lane) * 64;
    for (int k = 0; k < 64; k++) w_all[base + k] = row[k] * inv;
  }
}

// ---------------------------------------------------------------------------
// Phase 2 helpers: async prefetch of one step's operands into LDS stage.
// w_t: 4 rows x 256B (64 b128 chunks); emb_t: 4 rows x 512B (128 chunks,
// gathered through concept_ids); correctness: 4 b32.  Issued by ONE wave.
// ---------------------------------------------------------------------------
__device__ __forceinline__ void prefetch_step(
    int tn, int bufdst, int lane, int b0,
    const int* __restrict__ concept_ids, const float* __restrict__ correctness,
    const float* __restrict__ concept_embed, const float* __restrict__ w_all,
    unsigned wst_off, unsigned est_off, unsigned cst_off) {
  int cidv = 0;
  if (lane < 4) cidv = concept_ids[(size_t)(b0 + lane) * Ssz + tn];
  unsigned wbase = wst_off + (unsigned)bufdst * 1024u;
#pragma unroll
  for (int rep = 0; rep < 2; rep++) {
    int ci = rep * 32 + lane;               // 0..63: j = ci>>4, chunk = ci&15
    const char* g = (const char*)w_all +
        ((((size_t)(b0 + (ci >> 4)) * Ssz + tn) * 64) + (size_t)(ci & 15) * 4) * 4;
    async_copy_b128(wbase + (unsigned)ci * 16u, g);
  }
  unsigned ebase = est_off + (unsigned)bufdst * 2048u;
#pragma unroll
  for (int rep = 0; rep < 4; rep++) {
    int ci = rep * 32 + lane;               // 0..127: j = ci>>5, chunk = ci&31
    int cid = __shfl(cidv, ci >> 5, 32);
    const char* g = (const char*)concept_embed + (size_t)cid * 512 + (size_t)(ci & 31) * 16;
    async_copy_b128(ebase + (unsigned)ci * 16u, g);
  }
  if (lane < 4) {
    const char* g = (const char*)correctness + ((size_t)(b0 + lane) * Ssz + tn) * 4;
    async_copy_b32(cst_off + (unsigned)bufdst * 64u + (unsigned)lane * 4u, g);
  }
}

// ---------------------------------------------------------------------------
// Phase 2: 200-step scan.  512 threads (16 waves), 4 batches/block.
// vm LDS-resident (TDM-initialized); per-step operands async-double-buffered;
// erase/add/summary via WMMA with register-cached bf16 weight tiles.
// ---------------------------------------------------------------------------
__global__ void __launch_bounds__(512) dkvmn_scan(
    const int* __restrict__ concept_ids, const float* __restrict__ correctness,
    const float* __restrict__ concept_embed, const float* __restrict__ init_vm,
    const float* __restrict__ w_all,
    const __bf16* __restrict__ We_sw, const __bf16* __restrict__ Wa_sw,
    const __bf16* __restrict__ Ws_sw,
    const float* __restrict__ be, const float* __restrict__ ba,
    const float* __restrict__ bs, const float* __restrict__ Wo,
    const float* __restrict__ bo,
    float* __restrict__ preds_out, float* __restrict__ vm_out) {
  extern __shared__ char smem_raw[];
  float*  vm       = (float*)smem_raw;          // [4][64][256]           65536 f
  float*  readbuf  = vm + 65536;                // [4][256]
  float*  erasebuf = readbuf + 1024;            // [4][256]
  float*  addbuf   = erasebuf + 1024;           // [4][256]
  float*  sumP     = addbuf + 1024;             // [2][4][128]
  float*  summary  = sumP + 1024;               // [4][128]
  float*  wst      = summary + 512;             // [2][4][64]  staged w_t
  float*  est      = wst + 512;                 // [2][4][128] staged emb_t
  float*  cst      = est + 1024;                // [2][16]     staged c_t
  __bf16* uA       = (__bf16*)(cst + 32);       // [16][256] A-src for We/Wa
  __bf16* Xbf      = uA + 16 * 256;             // [16][384] A-src for Ws

  const int tid = threadIdx.x, lane = tid & 31, wv = tid >> 5;  // wv 0..15
  const int b0 = blockIdx.x * BPB;
  const unsigned vm_off  = (unsigned)(uintptr_t)vm;
  const unsigned wst_off = (unsigned)(uintptr_t)wst;
  const unsigned est_off = (unsigned)(uintptr_t)est;
  const unsigned cst_off = (unsigned)(uintptr_t)cst;

  // ---- init: TDM broadcast of init_value_memory into the 4 vm slots ----
  if (wv == 0) {
#pragma unroll
    for (int j = 0; j < BPB; j++)
      tdm_1d_f32(vm_off + (unsigned)j * 65536u, init_vm, 16384u, false);
    __builtin_amdgcn_s_wait_tensorcnt(0);
  }
  for (int i = tid; i < 16 * 256; i += 512) uA[i] = (__bf16)0.f;
  for (int i = tid; i < 16 * 384; i += 512) Xbf[i] = (__bf16)0.f;

  // ---- register-cache this wave's weight tiles for the whole scan ----
  v16bf wgt[2][8];
  float biasAB[2];
  bool  isAdd[2];
#pragma unroll
  for (int jj = 0; jj < 2; jj++) {
    int g = 2 * wv + jj;                        // 0..31: We nt0..15, Wa nt0..15
    isAdd[jj] = (g >= 16);
    int nt = g & 15;
    const __bf16* src = isAdd[jj] ? Wa_sw : We_sw;
    biasAB[jj] = (isAdd[jj] ? ba : be)[nt * 16 + (lane & 15)];
#pragma unroll
    for (int kt = 0; kt < 8; kt++) wgt[jj][kt] = glb_b_tile(src, nt * 8 + kt, lane);
  }
  v16bf wgtS[6];                                // split-K summary tiles
  const int ntS = wv & 7, half = (wv >> 3) & 1, ks0 = half ? 6 : 0;
#pragma unroll
  for (int s = 0; s < 6; s++) wgtS[s] = glb_b_tile(Ws_sw, ntS * 12 + (ks0 + s), lane);

  // ---- async prefetch of step 0 operands into buffer 0 ----
  if (wv == 15) {
    prefetch_step(0, 0, lane, b0, concept_ids, correctness, concept_embed,
                  w_all, wst_off, est_off, cst_off);
    wait_async0();
  }
  __syncthreads();

  for (int t = 0; t < Ssz; t++) {
    const int buf = t & 1;
    const float* wstb = wst + buf * 256;
    const float* estb = est + buf * 512;
    const float* cstb = cst + buf * 16;

    // ---- P1: emb stage -> Xbf bf16 ; pred(t-1) ; issue prefetch(t+1) ----
    {
      int j = tid >> 7, d = tid & 127;
      Xbf[j * 384 + 256 + d] = (__bf16)estb[j * 128 + d];
    }
    if (t > 0 && tid < BPB) {
      float s = bo[0];
      for (int d = 0; d < 128; d++) s += summary[tid * 128 + d] * Wo[d];
      preds_out[(size_t)(b0 + tid) * Ssz + (t - 1)] = sigmoidf_(s);
    }
    if (wv == 15 && t + 1 < Ssz)
      prefetch_step(t + 1, buf ^ 1, lane, b0, concept_ids, correctness,
                    concept_embed, w_all, wst_off, est_off, cst_off);
    __syncthreads();

    // ---- P2: read = w_t . vm ; u = read*c -> uA ; read -> Xbf ----
    for (int i = tid; i < BPB * 256; i += 512) {
      int j = i >> 8, v = i & 255;
      const float* vmj = vm + j * 16384 + v;
      const float* wj = wstb + j * 64;
      float acc = 0.f;
#pragma unroll 8
      for (int k = 0; k < 64; k++) acc += wj[k] * vmj[k * 256];
      readbuf[i] = acc;
      uA[j * 256 + v] = (__bf16)(acc * cstb[j]);
      Xbf[j * 384 + v] = (__bf16)acc;
    }
    __syncthreads();

    // ---- P3: WMMA — erase/add (u@We, u@Wa) and summary partial (X@Ws) ----
#pragma unroll
    for (int jj = 0; jj < 2; jj++) {
      v8f acc = {};
#pragma unroll
      for (int kt = 0; kt < 8; kt++) {
        v16bf a = lds_a_tile(uA, 256, kt, lane);
        acc = wmma_bf16(a, wgt[jj][kt], acc);
      }
      int g = 2 * wv + jj;
      int col = (g & 15) * 16 + (lane & 15);
      int rb = (lane & 16) ? 8 : 0;
      float* outb = isAdd[jj] ? addbuf : erasebuf;
#pragma unroll
      for (int vr = 0; vr < 8; vr++) {
        int r = vr + rb;
        if (r < BPB) {
          float x = acc[vr] + biasAB[jj];
          outb[r * 256 + col] = isAdd[jj] ? tanhf(x) : sigmoidf_(x);
        }
      }
    }
    {
      v8f acc = {};
#pragma unroll
      for (int s = 0; s < 6; s++) {
        v16bf a = lds_a_tile(Xbf, 384, ks0 + s, lane);
        acc = wmma_bf16(a, wgtS[s], acc);
      }
      int col = ntS * 16 + (lane & 15);
      int rb = (lane & 16) ? 8 : 0;
#pragma unroll
      for (int vr = 0; vr < 8; vr++) {
        int r = vr + rb;
        if (r < BPB) sumP[(half * BPB + r) * 128 + col] = acc[vr];
      }
    }
    __syncthreads();

    // ---- P4: summary = tanh(p0+p1+bs) ; vm update (float4 packed) ----
    if (tid < BPB * 128) {
      int j = tid >> 7, d = tid & 127;
      summary[tid] = tanhf(sumP[j * 128 + d] + sumP[(BPB + j) * 128 + d] + bs[d]);
    }
    for (int i = tid; i < BPB * 64 * 256 / 4; i += 512) {
      int j = i >> 12, rem = i & 4095;
      int k = rem >> 6, v = (rem & 63) * 4;
      float wjk = wstb[j * 64 + k];
      float4 e = *(const float4*)&erasebuf[j * 256 + v];
      float4 a = *(const float4*)&addbuf[j * 256 + v];
      float4* pv = (float4*)&vm[j * 16384 + k * 256 + v];
      float4 m = *pv;
      m.x = m.x * (1.f - wjk * e.x) + wjk * a.x;
      m.y = m.y * (1.f - wjk * e.y) + wjk * a.y;
      m.z = m.z * (1.f - wjk * e.z) + wjk * a.z;
      m.w = m.w * (1.f - wjk * e.w) + wjk * a.w;
      *pv = m;
    }
    if (wv == 15) wait_async0();   // publish prefetched t+1 stage at barrier
    __syncthreads();
  }

  // ---- final pred (t = S-1) and TDM vm writeout ----
  if (tid < BPB) {
    float s = bo[0];
    for (int d = 0; d < 128; d++) s += summary[tid * 128 + d] * Wo[d];
    preds_out[(size_t)(b0 + tid) * Ssz + (Ssz - 1)] = sigmoidf_(s);
  }
  if (wv == 0) {
#pragma unroll
    for (int j = 0; j < BPB; j++)
      tdm_1d_f32(vm_off + (unsigned)j * 65536u,
                 vm_out + (size_t)(b0 + j) * 16384, 16384u, true);
    __builtin_amdgcn_s_wait_tensorcnt(0);
  }
}

// ---------------------------------------------------------------------------
extern "C" void kernel_launch(void* const* d_in, const int* in_sizes, int n_in,
                              void* d_out, int out_size, void* d_ws, size_t ws_size,
                              hipStream_t stream) {
  const int*   concept_ids   = (const int*)d_in[0];
  const float* correctness   = (const float*)d_in[1];
  const float* key_memory    = (const float*)d_in[2];
  const float* concept_embed = (const float*)d_in[3];
  const float* init_vm       = (const float*)d_in[4];
  const float* Wq = (const float*)d_in[5];  const float* bq = (const float*)d_in[6];
  const float* We = (const float*)d_in[7];  const float* be = (const float*)d_in[8];
  const float* Wa = (const float*)d_in[9];  const float* ba = (const float*)d_in[10];
  const float* Ws = (const float*)d_in[11]; const float* bs = (const float*)d_in[12];
  const float* Wo = (const float*)d_in[13]; const float* bo = (const float*)d_in[14];

  char* ws = (char*)d_ws;
  __bf16* Wq_sw = (__bf16*)(ws);            //  65536 B
  __bf16* KT_sw = (__bf16*)(ws + 65536);    //  32768 B
  __bf16* We_sw = (__bf16*)(ws + 98304);    // 131072 B
  __bf16* Wa_sw = (__bf16*)(ws + 229376);   // 131072 B
  __bf16* Ws_sw = (__bf16*)(ws + 360448);   //  98304 B
  float*  w_all = (float*)(ws + 458752);    // B*S*K f32 = 104857600 B

  float* preds = (float*)d_out;             // (B,S)
  float* vmout = preds + (size_t)Bsz * Ssz; // (B,K,V)

  // Phase 0: weight swizzle (229376 elements)
  dkvmn_prep<<<896, 256, 0, stream>>>(Wq, key_memory, We, Wa, Ws,
                                      Wq_sw, KT_sw, We_sw, Wa_sw, Ws_sw);

  // Phase 1: attention weights.  64 rows/block -> 6400 blocks.
  size_t lds1 = (size_t)(4 * 16 * 128 + 4 * 16 * 256) * 2 + (size_t)4 * 16 * 64 * 4; // 64KB
  hipFuncSetAttribute((const void*)dkvmn_attn,
                      hipFuncAttributeMaxDynamicSharedMemorySize, (int)lds1);
  dkvmn_attn<<<6400, 128, lds1, stream>>>(concept_ids, concept_embed,
                                          Wq_sw, KT_sw, bq, w_all);

  // Phase 2: scan.  4 batches/block -> 512 blocks, 307328 B LDS (< 320KB/WGP).
  size_t lds2 = (size_t)(65536 + 1024 + 1024 + 1024 + 1024 + 512 + 512 + 1024 + 32) * 4 +
                (size_t)(16 * 256 + 16 * 384) * 2;
  hipFuncSetAttribute((const void*)dkvmn_scan,
                      hipFuncAttributeMaxDynamicSharedMemorySize, (int)lds2);
  dkvmn_scan<<<Bsz / BPB, 512, lds2, stream>>>(
      concept_ids, correctness, concept_embed, init_vm, w_all,
      We_sw, Wa_sw, Ws_sw, be, ba, bs, Wo, bo, preds, vmout);
}
